// ViewPointFactory_14602888806851
// MI455X (gfx1250) — compile-verified, matrix operand order checked by reading, not run on hardware
//
#include <hip/hip_runtime.h>
#include <hip/hip_bf16.h>

typedef __bf16 bf16;
typedef __attribute__((ext_vector_type(16))) __bf16 v16bf;
typedef __attribute__((ext_vector_type(8)))  __bf16 v8bf;
typedef __attribute__((ext_vector_type(8)))  float  v8f;

#define B_     2
#define L_     1024
#define DQ_    1024
#define DSRC_  512
#define VIEWS_ 8
#define DVP_   128
#define WIDTH_ 4
#define TOPK_  8
#define NFLAT_ (VIEWS_ * DVP_)   // 1024

// Build a v16bf A/B fragment per CDNA5 16-bit WMMA layout from a contiguous
// 32-element (K-dim) bf16 row. sel = lane>=16.
// lane<16 : elements 0..7 = K0..7   (VGPR0..3), 8..15 = K16..23 (VGPR4..7)
// lane>=16: elements 0..7 = K8..15, 8..15 = K24..31
__device__ __forceinline__ v16bf load_frag(const bf16* row32, int sel) {
    v8bf lo = *(const v8bf*)(row32 + sel * 8);
    v8bf hi = *(const v8bf*)(row32 + sel * 8 + 16);
    return __builtin_shufflevector(lo, hi, 0, 1, 2, 3, 4, 5, 6, 7,
                                           8, 9, 10, 11, 12, 13, 14, 15);
}

__device__ __forceinline__ void load_cvt8(const float* __restrict__ p, v8bf& d) {
#pragma unroll
    for (int i = 0; i < 8; ++i) d[i] = (bf16)p[i];
}

// ---------------------------------------------------------------------------
// Kernel 1: projection GEMM.  out[m][n] = sum_d in[m][d]*wt[n][d] + bias[n]
// M = B*L = 2048, N = VIEWS*DVP = 1024, K = DQ = 1024.  Output bf16.
// Block: 256 threads = 8 waves, block tile 64(M) x 64(N), wave tile 16x32.
// LDS double-buffered: one barrier per K-step; next tile's global loads +
// bf16 converts overlap the current step's WMMAs.
// ---------------------------------------------------------------------------
__global__ __launch_bounds__(256)
void proj_gemm_kernel(const float* __restrict__ in,
                      const float* __restrict__ wt,
                      const float* __restrict__ bias,
                      bf16* __restrict__ out) {
    __shared__ __align__(16) bf16 At[2][64][32];
    __shared__ __align__(16) bf16 Bt[2][64][32];

    const int tid  = threadIdx.x;
    const int lane = tid & 31;
    const int wv   = tid >> 5;     // 0..7
    const int mt   = wv & 3;       // 0..3 -> 16-row tile within block
    const int nh   = wv >> 2;      // 0..1 -> 32-col half within block
    const int lsel = lane >> 4;
    const int lm   = lane & 15;

    const int blockM = blockIdx.y * 64;
    const int blockN = blockIdx.x * 64;

    const int lrow = tid >> 2;          // 0..63
    const int lcol = (tid & 3) * 8;     // 0,8,16,24

    const float* ap0 = in + (size_t)(blockM + lrow) * DQ_ + lcol;
    const float* bp0 = wt + (size_t)(blockN + lrow) * DQ_ + lcol;

    v8f acc0 = {};
    v8f acc1 = {};

    // Prologue: fetch K-step 0 into registers.
    v8bf av, bv;
    load_cvt8(ap0, av);
    load_cvt8(bp0, bv);

    const int NSTEP = DQ_ / 32;  // 32
    for (int ks = 0; ks < NSTEP; ++ks) {
        const int cur = ks & 1;
        *(v8bf*)(&At[cur][lrow][lcol]) = av;
        *(v8bf*)(&Bt[cur][lrow][lcol]) = bv;
        __syncthreads();

        // Prefetch next K-step while WMMAs run out of buffer `cur`.
        if (ks + 1 < NSTEP) {
            load_cvt8(ap0 + (ks + 1) * 32, av);
            load_cvt8(bp0 + (ks + 1) * 32, bv);
        }

        v16bf a  = load_frag(&At[cur][mt * 16 + lm][0], lsel);
        v16bf b0 = load_frag(&Bt[cur][nh * 32 + lm][0], lsel);
        v16bf b1 = load_frag(&Bt[cur][nh * 32 + 16 + lm][0], lsel);

        acc0 = __builtin_amdgcn_wmma_f32_16x16x32_bf16(
            false, a, false, b0, (short)0, acc0, false, false);
        acc1 = __builtin_amdgcn_wmma_f32_16x16x32_bf16(
            false, a, false, b1, (short)0, acc1, false, false);
        // No second barrier: double buffering guarantees buffer `cur^1`
        // is only rewritten after the next iteration's barrier.
    }

    // Epilogue: bias add + bf16 store.
    const int n0 = blockN + nh * 32 + lm;
    const int n1 = n0 + 16;
    const float bv0 = bias[n0];
    const float bv1 = bias[n1];
    const int mbase = blockM + mt * 16 + lsel * 8;
#pragma unroll
    for (int r = 0; r < 8; ++r) {
        out[(size_t)(mbase + r) * NFLAT_ + n0] = (bf16)(acc0[r] + bv0);
        out[(size_t)(mbase + r) * NFLAT_ + n1] = (bf16)(acc1[r] + bv1);
    }
}

// ---------------------------------------------------------------------------
// Kernel 2: score = Q.K^T per (b,v), 16 query rows per block; top-8 + softmax.
// Block: 128 threads (4 waves). Each wave sweeps 256 key columns (16 tiles).
// A fragments live in registers; B fragments register-double-buffered so the
// next tile's global loads overlap the current tile's WMMAs; t+2 rows are
// warmed with global_prefetch. Scores staged to 64KB LDS scoreboard, then
// per-row top-8 reduction + softmax.
// ---------------------------------------------------------------------------
__global__ __launch_bounds__(128)
void score_topk_kernel(const bf16* __restrict__ Qp,
                       const bf16* __restrict__ Kp,
                       float* __restrict__ wout,
                       int* __restrict__ iout) {
    __shared__ __align__(16) float sc[16][L_];     // 64 KB
    __shared__ float cv[16][64];
    __shared__ int   ci[16][64];

    const int tid  = threadIdx.x;
    const int lane = tid & 31;
    const int wv   = tid >> 5;      // 0..3
    const int lsel = lane >> 4;
    const int lm   = lane & 15;

    const int bv = blockIdx.y;      // b*VIEWS + v
    const int b  = bv >> 3;
    const int v  = bv & 7;
    const int q0 = blockIdx.x * 16;

    // A fragments: 16 query rows x 128 K, kept in registers (4 x v16bf).
    const bf16* qrow =
        Qp + (size_t)(b * L_ + q0 + lm) * NFLAT_ + v * DVP_;
    v16bf afrag[4];
#pragma unroll
    for (int kk = 0; kk < 4; ++kk)
        afrag[kk] = load_frag(qrow + kk * 32, lsel);

    const bf16* kbase = Kp + (size_t)(b * L_) * NFLAT_ + v * DVP_;
    const bf16* krow0 = kbase + (size_t)(wv * 256 + lm) * NFLAT_;

    // Register double-buffer for B fragments.
    v16bf bfr[2][4];
#pragma unroll
    for (int kk = 0; kk < 4; ++kk)
        bfr[0][kk] = load_frag(krow0 + kk * 32, lsel);

#pragma unroll
    for (int t = 0; t < 16; ++t) {
        if (t + 2 < 16)
            __builtin_prefetch(krow0 + (size_t)(t + 2) * 16 * NFLAT_, 0, 1);
        if (t + 1 < 16) {
            const bf16* kr = krow0 + (size_t)(t + 1) * 16 * NFLAT_;
#pragma unroll
            for (int kk = 0; kk < 4; ++kk)
                bfr[(t + 1) & 1][kk] = load_frag(kr + kk * 32, lsel);
        }
        v8f acc = {};
#pragma unroll
        for (int kk = 0; kk < 4; ++kk) {
            acc = __builtin_amdgcn_wmma_f32_16x16x32_bf16(
                false, afrag[kk], false, bfr[t & 1][kk], (short)0, acc,
                false, false);
        }
        const int n0 = wv * 256 + t * 16;
#pragma unroll
        for (int r = 0; r < 8; ++r)
            sc[lsel * 8 + r][n0 + lm] = acc[r];
    }
    __syncthreads();

    // Partial top-8: 8 threads per row, 128 columns each.
    const int row = tid >> 3;
    const int seg = tid & 7;
    float tv[TOPK_];
    int   tix[TOPK_];
#pragma unroll
    for (int i = 0; i < TOPK_; ++i) { tv[i] = -3.0e38f; tix[i] = 0; }
    for (int j = seg * 128; j < seg * 128 + 128; ++j) {
        float s = sc[row][j];
        if (s > tv[TOPK_ - 1]) {
            int p = TOPK_ - 1;
            while (p > 0 && tv[p - 1] < s) {
                tv[p] = tv[p - 1]; tix[p] = tix[p - 1]; --p;
            }
            tv[p] = s; tix[p] = j;
        }
    }
#pragma unroll
    for (int i = 0; i < TOPK_; ++i) {
        cv[row][seg * 8 + i] = tv[i];
        ci[row][seg * 8 + i] = tix[i];
    }
    __syncthreads();

    if (seg == 0) {
        float fv[TOPK_];
        int   fix[TOPK_];
#pragma unroll
        for (int i = 0; i < TOPK_; ++i) { fv[i] = -3.0e38f; fix[i] = 0; }
        for (int j = 0; j < 64; ++j) {
            float s  = cv[row][j];
            int   ix = ci[row][j];
            if (s > fv[TOPK_ - 1]) {
                int p = TOPK_ - 1;
                while (p > 0 && fv[p - 1] < s) {
                    fv[p] = fv[p - 1]; fix[p] = fix[p - 1]; --p;
                }
                fv[p] = s; fix[p] = ix;
            }
        }
        // softmax over the top-8 (fv[0] is the max).
        const float m = fv[0];
        float e[TOPK_], sum = 0.f;
#pragma unroll
        for (int i = 0; i < TOPK_; ++i) { e[i] = __expf(fv[i] - m); sum += e[i]; }
        const float inv = 1.f / sum;
        const size_t off = ((size_t)bv * L_ + q0 + row) * TOPK_;
#pragma unroll
        for (int i = 0; i < TOPK_; ++i) {
            wout[off + i] = e[i] * inv;
            iout[off + i] = fix[i];
        }
    }
}

// ---------------------------------------------------------------------------
// Kernel 3: out[b,v,q,w,d] = sum_j weights[j] * source[b, idx_j + w - 1, d]
// One block per (b,v,q) row; 2048 fp32 outputs, coalesced along d.
// ---------------------------------------------------------------------------
__global__ __launch_bounds__(256)
void gather_out_kernel(const float* __restrict__ src,
                       const float* __restrict__ wts,
                       const int* __restrict__ idx,
                       float* __restrict__ out) {
    __shared__ float w8[TOPK_];
    __shared__ int   i8[TOPK_];

    const int tid = threadIdx.x;
    const int g   = blockIdx.x;          // bv*L + q
    const int bv  = g >> 10;
    const int b   = bv >> 3;

    if (tid < TOPK_) {
        w8[tid] = wts[(size_t)g * TOPK_ + tid];
        i8[tid] = idx[(size_t)g * TOPK_ + tid];
    }
    __syncthreads();

    const float* sb = src + (size_t)b * L_ * DSRC_;
    float* ob = out + (size_t)g * WIDTH_ * DSRC_;

#pragma unroll
    for (int s = 0; s < (WIDTH_ * DSRC_) / 256; ++s) {
        const int e = tid + s * 256;
        const int w = e >> 9;            // / DSRC_
        const int d = e & (DSRC_ - 1);
        float acc = 0.f;
#pragma unroll
        for (int j = 0; j < TOPK_; ++j) {
            const int r = i8[j] + w - 1;     // PAD_F = 1
            if (r >= 0 && r < L_) acc += w8[j] * sb[(size_t)r * DSRC_ + d];
        }
        ob[e] = acc;
    }
}

extern "C" void kernel_launch(void* const* d_in, const int* in_sizes, int n_in,
                              void* d_out, int out_size, void* d_ws, size_t ws_size,
                              hipStream_t stream) {
    const float* query  = (const float*)d_in[0];
    const float* key_t  = (const float*)d_in[1];
    const float* source = (const float*)d_in[2];
    const float* wq     = (const float*)d_in[3];
    const float* bq     = (const float*)d_in[4];
    const float* wk     = (const float*)d_in[5];
    const float* bk     = (const float*)d_in[6];
    float* out = (float*)d_out;

    char* ws = (char*)d_ws;
    bf16*  Qp  = (bf16*)ws;                                   // 4 MB
    bf16*  Kp  = (bf16*)(ws + ((size_t)4 << 20));             // 4 MB
    float* wts = (float*)(ws + ((size_t)8 << 20));            // 512 KB
    int*   idx = (int*)(ws + ((size_t)8 << 20) + ((size_t)1 << 19));

    dim3 g1(NFLAT_ / 64, (B_ * L_) / 64, 1);                  // 16 x 32
    proj_gemm_kernel<<<g1, 256, 0, stream>>>(query, wq, bq, Qp);
    proj_gemm_kernel<<<g1, 256, 0, stream>>>(key_t, wk, bk, Kp);

    dim3 g2(L_ / 16, B_ * VIEWS_, 1);                         // 64 x 16
    score_topk_kernel<<<g2, 128, 0, stream>>>(Qp, Kp, wts, idx);

    gather_out_kernel<<<B_ * VIEWS_ * L_, 256, 0, stream>>>(source, wts, idx, out);
}
